// D_R_30863634989335
// MI455X (gfx1250) — matrix operand order repaired and checked by reading, not compiled, not run on hardware
//
#include <hip/hip_runtime.h>
#include <hip/hip_bf16.h>

// ---------------------------------------------------------------------------
// Problem constants (match reference)
// ---------------------------------------------------------------------------
#define B_DIM 128
#define D_DIM 64
#define L_DIM 512
#define H_DIM 512
#define K_IMF 4
#define M_BR  16
#define NEG_SLOPE 0.01f
#define EPS_BN 1e-5f

typedef __attribute__((ext_vector_type(16))) _Float16 v16h;
typedef __attribute__((ext_vector_type(8)))  float    v8f;

__device__ __forceinline__ float leaky_f(float v) { return v >= 0.f ? v : NEG_SLOPE * v; }

// ---------------------------------------------------------------------------
// WMMA GEMM (f16 operands, f32 accumulate):
//   Y[M,N] = act( X[M,K] * W[N,K]^T + bias[N] )
// X row stride lda (halves), W rows contiguous (stride K), Y row stride ldy.
// Requires M%128==0, N%64==0, K%32==0 (true at every call site).
// 256 threads = 8 waves; block tile 128x64x32; wave tile 32x32 = 2x2 WMMA.
// LDS is double-buffered and filled with GLOBAL_LOAD_ASYNC_TO_LDS_B128
// (ASYNCcnt-tracked, L2->LDS direct) so stage k+1 streams in during the
// WMMAs on stage k.  mode: bit0 = leaky, bit1 = write f16 to Yh (else f32 Yf).
// ---------------------------------------------------------------------------
#define GEMM_BM 128
#define GEMM_BN 64
#define GEMM_BK 32
#define AS_STR  40   // padded LDS row stride (halves)
#define BS_STR  40

__global__ __launch_bounds__(256, 2)
void gemm_h_wmma(const _Float16* __restrict__ X, long lda,
                 const _Float16* __restrict__ W,
                 const float* __restrict__ bias,
                 float* __restrict__ Yf, _Float16* __restrict__ Yh, long ldy,
                 int Kd, int mode)
{
  __shared__ _Float16 As[2][GEMM_BM * AS_STR];
  __shared__ _Float16 Bs[2][GEMM_BN * BS_STR];

  const int t    = threadIdx.x;
  const int lane = t & 31;
  const int wave = t >> 5;
  const int wm   = wave & 3;                    // 4 waves along M (32 rows each)
  const int wn   = wave >> 2;                   // 2 waves along N (32 cols each)
  const long bm  = (long)blockIdx.y * GEMM_BM;
  const long bn  = (long)blockIdx.x * GEMM_BN;

  // staging maps: A = 128x32 halves (16/thread = 2x B128), B = 64x32 (8/thread = 1x B128)
  const int arow = t >> 1;
  const int acol = (t & 1) << 4;
  const int brow = t >> 2;
  const int bcol = (t & 3) << 3;

  const _Float16* aptr = X + (bm + arow) * lda + acol;
  const _Float16* bptr = W + (long)(bn + brow) * Kd + bcol;

  // LDS byte addresses for this thread's async destinations (low 32 bits of
  // a generic LDS pointer are the LDS offset).
  const unsigned ldsA[2] = {
    (unsigned)(unsigned long)&As[0][arow * AS_STR + acol],
    (unsigned)(unsigned long)&As[1][arow * AS_STR + acol] };
  const unsigned ldsB[2] = {
    (unsigned)(unsigned long)&Bs[0][brow * BS_STR + bcol],
    (unsigned)(unsigned long)&Bs[1][brow * BS_STR + bcol] };

  auto issue_stage = [&](int buf, int k0) {
    const _Float16* ga = aptr + k0;
    const _Float16* gb = bptr + k0;
    const unsigned la = ldsA[buf];
    const unsigned lb = ldsB[buf];
    asm volatile("global_load_async_to_lds_b128 %0, %1, off"
                 :: "v"(la), "v"(ga) : "memory");
    asm volatile("global_load_async_to_lds_b128 %0, %1, off"
                 :: "v"(la + 16u), "v"(ga + 8) : "memory");
    asm volatile("global_load_async_to_lds_b128 %0, %1, off"
                 :: "v"(lb), "v"(gb) : "memory");
  };

  union CF { v8f v; float f[8]; };
  CF acc[2][2];
#pragma unroll
  for (int i = 0; i < 2; ++i)
#pragma unroll
    for (int j = 0; j < 2; ++j)
#pragma unroll
      for (int v = 0; v < 8; ++v) acc[i][j].f[v] = 0.f;

  const int nk = Kd / GEMM_BK;
  issue_stage(0, 0);                                   // stage 0
  if (nk > 1) issue_stage(1, GEMM_BK);                 // stage 1 in flight

  for (int s = 0; s < nk; ++s) {
    // In-order ASYNCcnt: <=3 outstanding means stage s has fully landed while
    // stage s+1 (3 ops) may still be streaming.
    if (s + 1 < nk) asm volatile("s_wait_asynccnt 0x3" ::: "memory");
    else            asm volatile("s_wait_asynccnt 0x0" ::: "memory");
    __syncthreads();                                   // all waves' copies visible

    const int buf = s & 1;
    union AF { v16h v; _Float16 h[16]; };
    AF af[2], bf[2];
    const int l15 = lane & 15;
    const int akb = (lane < 16) ? 0 : 16;              // A: lanes16-31 hold K 16..31
#pragma unroll
    for (int tm = 0; tm < 2; ++tm) {
      const _Float16* ap = &As[buf][(wm * 32 + tm * 16 + l15) * AS_STR];
#pragma unroll
      for (int e = 0; e < 8; ++e)  af[tm].h[e] = ap[akb + e];
#pragma unroll
      for (int e = 8; e < 16; ++e) af[tm].h[e] = ap[akb + e];
    }
    const int bkb = (lane < 16) ? 0 : 16;              // B: lanes16-31 hold K 16..31
#pragma unroll
    for (int tn = 0; tn < 2; ++tn) {
      const _Float16* bp = &Bs[buf][(wn * 32 + tn * 16 + l15) * BS_STR + bkb];
#pragma unroll
      for (int e = 0; e < 16; ++e) bf[tn].h[e] = bp[e];
    }
#pragma unroll
    for (int tm = 0; tm < 2; ++tm)
#pragma unroll
      for (int tn = 0; tn < 2; ++tn)
        acc[tm][tn].v = __builtin_amdgcn_wmma_f32_16x16x32_f16(
            false, af[tm].v, false, bf[tn].v, (short)0, acc[tm][tn].v, false, false);
    __syncthreads();                                   // everyone done reading buf
    if (s + 2 < nk) issue_stage(buf, (s + 2) * GEMM_BK);
  }

  // Epilogue: C/D layout — lane<16 -> M=v, lane>=16 -> M=8+v, N=lane&15
  const int mo = (lane < 16) ? 0 : 8;
  const bool do_leaky = (mode & 1) != 0;
  const bool to_half  = (mode & 2) != 0;
#pragma unroll
  for (int tm = 0; tm < 2; ++tm) {
#pragma unroll
    for (int tn = 0; tn < 2; ++tn) {
      const long c = bn + wn * 32 + tn * 16 + (lane & 15);
      const float bv = bias ? bias[c] : 0.f;
#pragma unroll
      for (int v = 0; v < 8; ++v) {
        const long r = bm + wm * 32 + tm * 16 + mo + v;
        float val = acc[tm][tn].f[v] + bv;
        if (do_leaky) val = leaky_f(val);
        if (to_half) Yh[r * ldy + c] = (_Float16)val;
        else         Yf[r * ldy + c] = val;
      }
    }
  }
}

// ---------------------------------------------------------------------------
// EMD-like decomposition: K-1 learned 3-tap smoothers + residual.
// One block per (b,d) row; single LDS buffer updated in place, with edge padding.
// Writes imf k to out*[(bd*branches + mbase + k)*L + l]; f32 and/or f16 out.
// ---------------------------------------------------------------------------
__global__ __launch_bounds__(256)
void mcd_kernel(const float* __restrict__ x, const float* __restrict__ w,
                float* __restrict__ outf, _Float16* __restrict__ outh,
                int branches, int mbase)
{
  __shared__ float r[L_DIM + 2];
  const int t = threadIdx.x;
  const long bd = blockIdx.x;
  const float* row = x + bd * L_DIM;
  r[t + 1]   = row[t];
  r[t + 257] = row[t + 256];
  if (t == 0) { r[0] = row[0]; r[L_DIM + 1] = row[L_DIM - 1]; }
  __syncthreads();
  const long obase = (bd * (long)branches + mbase) * L_DIM;
#pragma unroll
  for (int k = 0; k < K_IMF - 1; ++k) {
    const float w0 = w[k * 3 + 0], w1 = w[k * 3 + 1], w2 = w[k * 3 + 2];
    const float s0 = w0 * r[t]       + w1 * r[t + 1]   + w2 * r[t + 2];
    const float s1 = w0 * r[t + 256] + w1 * r[t + 257] + w2 * r[t + 258];
    const float i0 = r[t + 1] - s0, i1 = r[t + 257] - s1;
    const long o0 = obase + (long)k * L_DIM + t;
    if (outf) { outf[o0] = i0; outf[o0 + 256] = i1; }
    if (outh) { outh[o0] = (_Float16)i0; outh[o0 + 256] = (_Float16)i1; }
    __syncthreads();
    r[t + 1] = s0; r[t + 257] = s1;
    __syncthreads();
    if (t == 0) { r[0] = r[1]; r[L_DIM + 1] = r[L_DIM]; }
    __syncthreads();
  }
  const long oR = obase + (long)(K_IMF - 1) * L_DIM + t;
  if (outf) { outf[oR] = r[t + 1]; outf[oR + 256] = r[t + 257]; }
  if (outh) { outh[oR] = (_Float16)r[t + 1]; outh[oR + 256] = (_Float16)r[t + 257]; }
}

// ---------------------------------------------------------------------------
// BatchNorm (affine=False, batch stats) per channel D over [B, D, R] view.
// ---------------------------------------------------------------------------
__global__ void zero_kernel(float* p) { p[threadIdx.x] = 0.f; }

__global__ __launch_bounds__(256)
void bn_stats_kernel(const float* __restrict__ x, float* __restrict__ stats, long R)
{
  __shared__ float ss[256], sq[256];
  const long b = blockIdx.x, d = blockIdx.y;
  const float* p = x + (b * D_DIM + d) * R;
  float s = 0.f, q = 0.f;
  for (long i = threadIdx.x; i < R; i += 256) { float v = p[i]; s += v; q += v * v; }
  ss[threadIdx.x] = s; sq[threadIdx.x] = q;
  __syncthreads();
  for (int off = 128; off > 0; off >>= 1) {
    if (threadIdx.x < off) {
      ss[threadIdx.x] += ss[threadIdx.x + off];
      sq[threadIdx.x] += sq[threadIdx.x + off];
    }
    __syncthreads();
  }
  if (threadIdx.x == 0) {
    atomicAdd(&stats[d], ss[0]);
    atomicAdd(&stats[D_DIM + d], sq[0]);
  }
}

// Normalize + leaky in place; optional f16 shadow for downstream WMMA GEMMs.
__global__ __launch_bounds__(256)
void bn_apply_kernel(float* __restrict__ x, _Float16* __restrict__ shadow,
                     const float* __restrict__ stats, long R, float invcnt, long n)
{
  const long i = (long)blockIdx.x * 256 + threadIdx.x;
  if (i >= n) return;
  const int d = (int)((i / R) % D_DIM);
  const float m   = stats[d] * invcnt;
  const float var = stats[D_DIM + d] * invcnt - m * m;
  float v = (x[i] - m) * rsqrtf(var + EPS_BN);
  v = leaky_f(v);
  x[i] = v;
  if (shadow) shadow[i] = (_Float16)v;
}

// ---------------------------------------------------------------------------
// Gumbel-softmax branch selection: one thread per (b,d,k).
// ---------------------------------------------------------------------------
__device__ __forceinline__ unsigned hash_u32(unsigned x) {
  x ^= x >> 16; x *= 0x7feb352dU; x ^= x >> 15; x *= 0x846ca68bU; x ^= x >> 16;
  return x;
}
__device__ __forceinline__ float gumbel_of(unsigned seed, unsigned idx) {
  const unsigned h = hash_u32(seed ^ (idx * 0x9E3779B9U) ^ 0x85ebca6bU);
  const float u = ((float)h + 0.5f) * (1.0f / 4294967296.0f);
  return -logf(-logf(u));
}

__global__ __launch_bounds__(256)
void sel_kernel(const float* __restrict__ hbs, const float* __restrict__ w2,
                const float* __restrict__ b2, float* __restrict__ sel, unsigned seed)
{
  const int i = blockIdx.x * 256 + threadIdx.x;      // < B*D*K
  const float* h = hbs + (long)i * 64;
  float l0 = b2[0], l1 = b2[1];
#pragma unroll 8
  for (int j = 0; j < 64; ++j) { float hv = h[j]; l0 += hv * w2[j]; l1 += hv * w2[64 + j]; }
  l0 += gumbel_of(seed, 2u * (unsigned)i);
  l1 += gumbel_of(seed, 2u * (unsigned)i + 1u);
  const float mx = fmaxf(l0, l1);
  const float e0 = expf(l0 - mx), e1 = expf(l1 - mx);
  const float inv = 1.f / (e0 + e1);
  sel[2 * i]     = e0 * inv;
  sel[2 * i + 1] = e1 * inv;
}

// x_summed[...,c] = sum_k ximf[b,d,k,l] * sel[b,d,k,c]; f16 out (GEMM input only)
__global__ __launch_bounds__(256)
void combine_kernel(const float* __restrict__ ximf, const float* __restrict__ sel,
                    _Float16* __restrict__ x0, _Float16* __restrict__ x1)
{
  const long i = (long)blockIdx.x * 256 + threadIdx.x;   // < B*D*L
  const long bd = i >> 9;
  const int  l  = (int)(i & 511);
  const float* xb = ximf + (bd << 11) + l;               // bd*K*L
  const float* s  = sel + (bd << 3);                     // bd*K*2
  float a0 = 0.f, a1 = 0.f;
#pragma unroll
  for (int k = 0; k < K_IMF; ++k) {
    const float v = xb[(long)k << 9];
    a0 += v * s[2 * k];
    a1 += v * s[2 * k + 1];
  }
  x0[i] = (_Float16)a0; x1[i] = (_Float16)a1;
}

// f32 -> f16 weight conversion (once per launch)
__global__ __launch_bounds__(256)
void cvt_kernel(const float* __restrict__ src, _Float16* __restrict__ dst, long n)
{
  const long i = (long)blockIdx.x * 256 + threadIdx.x;
  if (i < n) dst[i] = (_Float16)src[i];
}

// ---------------------------------------------------------------------------
// Host-side orchestration
// ---------------------------------------------------------------------------
struct LvlP {
  const float *bw1, *bb1, *bw2, *bb2;
  const float *rlw1, *rlb1, *rlw2, *rlb2;
  const float *rrw1, *rrb1, *rrw2, *rrb2;
};
struct LvlH {      // f16 copies of the WMMA-consumed weights
  _Float16 *bw1, *rlw1, *rlw2, *rrw1, *rrw2;
};

static void launch_gemm(const _Float16* X, long lda, const _Float16* W, const float* bias,
                        float* Yf, _Float16* Yh, long ldy,
                        int M, int N, int K, int mode, hipStream_t s)
{
  dim3 grid((unsigned)(N / GEMM_BN), (unsigned)(M / GEMM_BM));
  gemm_h_wmma<<<grid, 256, 0, s>>>(X, lda, W, bias, Yf, Yh, ldy, K, mode);
}

static void run_bn(float* buf, _Float16* shadow, long R, float* stats, hipStream_t s)
{
  zero_kernel<<<1, 128, 0, s>>>(stats);
  dim3 g((unsigned)B_DIM, (unsigned)D_DIM);
  bn_stats_kernel<<<g, 256, 0, s>>>(buf, stats, R);
  const long n = (long)B_DIM * D_DIM * R;
  const float invcnt = 1.0f / (float)((long)B_DIM * R);
  bn_apply_kernel<<<(unsigned)((n + 255) / 256), 256, 0, s>>>(buf, shadow, stats, R, invcnt, n);
}

static void cvt(const float* src, _Float16* dst, long n, hipStream_t s)
{
  cvt_kernel<<<(unsigned)((n + 255) / 256), 256, 0, s>>>(src, dst, n);
}

// One internal tree node: mcd -> branch-select (gumbel) -> two recon towers.
static void process_internal(const float* xin, const float* mcdw,
                             const LvlP& P, const LvlH& Ph,
                             float* ximf, _Float16* ximf_h, float* hbs, float* sel,
                             _Float16* xsum0, _Float16* xsum1,
                             float* t1, _Float16* t1_h,
                             float* outL, float* outR, float* stats,
                             unsigned seed, hipStream_t s)
{
  const int BD = B_DIM * D_DIM;                   // 8192
  mcd_kernel<<<(unsigned)BD, 256, 0, s>>>(xin, mcdw, ximf, ximf_h, K_IMF, 0);

  // branch-select MLP: [B*D*K, L] x [64, L]^T
  launch_gemm(ximf_h, L_DIM, Ph.bw1, P.bb1, hbs, nullptr, 64, BD * K_IMF, 64, L_DIM, 0, s);
  run_bn(hbs, nullptr, (long)K_IMF * 64, stats, s);      // per-D over (B,K,64)
  sel_kernel<<<(unsigned)(BD * K_IMF / 256), 256, 0, s>>>(hbs, P.bw2, P.bb2, sel, seed);
  combine_kernel<<<(unsigned)((long)BD * L_DIM / 256), 256, 0, s>>>(ximf, sel, xsum0, xsum1);

  // left recon tower: Linear -> BN -> LeakyReLU -> Linear -> BN -> LeakyReLU
  launch_gemm(xsum0, L_DIM, Ph.rlw1, P.rlb1, t1, nullptr, H_DIM, BD, H_DIM, L_DIM, 0, s);
  run_bn(t1, t1_h, H_DIM, stats, s);
  launch_gemm(t1_h, H_DIM, Ph.rlw2, P.rlb2, outL, nullptr, L_DIM, BD, L_DIM, H_DIM, 0, s);
  run_bn(outL, nullptr, L_DIM, stats, s);
  // right recon tower
  launch_gemm(xsum1, L_DIM, Ph.rrw1, P.rrb1, t1, nullptr, H_DIM, BD, H_DIM, L_DIM, 0, s);
  run_bn(t1, t1_h, H_DIM, stats, s);
  launch_gemm(t1_h, H_DIM, Ph.rrw2, P.rrb2, outR, nullptr, L_DIM, BD, L_DIM, H_DIM, 0, s);
  run_bn(outR, nullptr, L_DIM, stats, s);
}

extern "C" void kernel_launch(void* const* d_in, const int* in_sizes, int n_in,
                              void* d_out, int out_size, void* d_ws, size_t ws_size,
                              hipStream_t stream)
{
  (void)in_sizes; (void)n_in; (void)out_size; (void)ws_size;

  const float* x    = (const float*)d_in[0];
  const float* mcd2 = (const float*)d_in[1];
  const float* mcd1 = (const float*)d_in[2];
  const float* mcd0 = (const float*)d_in[3];

  LvlP P2 = { (const float*)d_in[4],  (const float*)d_in[5],  (const float*)d_in[6],  (const float*)d_in[7],
              (const float*)d_in[8],  (const float*)d_in[9],  (const float*)d_in[10], (const float*)d_in[11],
              (const float*)d_in[12], (const float*)d_in[13], (const float*)d_in[14], (const float*)d_in[15] };
  LvlP P1 = { (const float*)d_in[16], (const float*)d_in[17], (const float*)d_in[18], (const float*)d_in[19],
              (const float*)d_in[20], (const float*)d_in[21], (const float*)d_in[22], (const float*)d_in[23],
              (const float*)d_in[24], (const float*)d_in[25], (const float*)d_in[26], (const float*)d_in[27] };
  const float* fc_w1 = (const float*)d_in[28];
  const float* fc_b1 = (const float*)d_in[29];
  const float* fc_w2 = (const float*)d_in[30];
  const float* fc_b2 = (const float*)d_in[31];

  // ---- Workspace layout (byte allocator, 256B aligned chunks) ----
  const long BD   = (long)B_DIM * D_DIM;            // 8192
  const long BDL  = BD * L_DIM;                     // 4,194,304
  const long WREC = (long)H_DIM * L_DIM;            // 262,144
  char* wsb = (char*)d_ws;
  auto alloc = [&](size_t bytes) -> void* {
    void* p = (void*)wsb; wsb += (bytes + 255) & ~(size_t)255; return p;
  };
  _Float16* total_h = (_Float16*)alloc(sizeof(_Float16) * BD * M_BR * L_DIM); // [B,D,16,L] f16
  float*    ximf    = (float*)   alloc(sizeof(float)    * BD * K_IMF * L_DIM);
  _Float16* ximf_h  = (_Float16*)alloc(sizeof(_Float16) * BD * K_IMF * L_DIM);
  float*    hbs     = (float*)   alloc(sizeof(float)    * BD * K_IMF * 64);
  float*    sel     = (float*)   alloc(sizeof(float)    * BD * K_IMF * 2);
  _Float16* xsum0   = (_Float16*)alloc(sizeof(_Float16) * BDL);
  _Float16* xsum1   = (_Float16*)alloc(sizeof(_Float16) * BDL);
  float*    nl      = (float*)   alloc(sizeof(float)    * BDL);
  float*    nr      = (float*)   alloc(sizeof(float)    * BDL);
  float*    cl      = (float*)   alloc(sizeof(float)    * BDL);
  float*    cr      = (float*)   alloc(sizeof(float)    * BDL);
  float*    t1      = (float*)   alloc(sizeof(float)    * BDL);
  _Float16* t1_h    = (_Float16*)alloc(sizeof(_Float16) * BDL);
  float*    stats   = (float*)   alloc(sizeof(float)    * 128);
  // f16 weights
  LvlH Ph2, Ph1;
  Ph2.bw1  = (_Float16*)alloc(sizeof(_Float16) * 64 * L_DIM);
  Ph2.rlw1 = (_Float16*)alloc(sizeof(_Float16) * WREC);
  Ph2.rlw2 = (_Float16*)alloc(sizeof(_Float16) * WREC);
  Ph2.rrw1 = (_Float16*)alloc(sizeof(_Float16) * WREC);
  Ph2.rrw2 = (_Float16*)alloc(sizeof(_Float16) * WREC);
  Ph1.bw1  = (_Float16*)alloc(sizeof(_Float16) * 64 * L_DIM);
  Ph1.rlw1 = (_Float16*)alloc(sizeof(_Float16) * WREC);
  Ph1.rlw2 = (_Float16*)alloc(sizeof(_Float16) * WREC);
  Ph1.rrw1 = (_Float16*)alloc(sizeof(_Float16) * WREC);
  Ph1.rrw2 = (_Float16*)alloc(sizeof(_Float16) * WREC);
  _Float16* fcw1_h = (_Float16*)alloc(sizeof(_Float16) * M_BR * WREC);
  _Float16* fcw2_h = (_Float16*)alloc(sizeof(_Float16) * M_BR * WREC);
  _Float16* hh_h   = ximf_h;   // FC hidden [8192,512]; ximf_h is dead by then

  // ---- One-time weight conversion to f16 ----
  cvt(P2.bw1,  Ph2.bw1,  64 * L_DIM, stream);
  cvt(P2.rlw1, Ph2.rlw1, WREC, stream); cvt(P2.rlw2, Ph2.rlw2, WREC, stream);
  cvt(P2.rrw1, Ph2.rrw1, WREC, stream); cvt(P2.rrw2, Ph2.rrw2, WREC, stream);
  cvt(P1.bw1,  Ph1.bw1,  64 * L_DIM, stream);
  cvt(P1.rlw1, Ph1.rlw1, WREC, stream); cvt(P1.rlw2, Ph1.rlw2, WREC, stream);
  cvt(P1.rrw1, Ph1.rrw1, WREC, stream); cvt(P1.rrw2, Ph1.rrw2, WREC, stream);
  cvt(fc_w1, fcw1_h, (long)M_BR * WREC, stream);
  cvt(fc_w2, fcw2_h, (long)M_BR * WREC, stream);

  // ---- Level 2 (root) ----
  process_internal(x, mcd2, P2, Ph2, ximf, ximf_h, hbs, sel, xsum0, xsum1,
                   t1, t1_h, nl, nr, stats, 0xA341u, stream);

  // ---- Level 1, left subtree -> branches 0..7 ----
  process_internal(nl, mcd1, P1, Ph1, ximf, ximf_h, hbs, sel, xsum0, xsum1,
                   t1, t1_h, cl, cr, stats, 0xB777u, stream);
  mcd_kernel<<<(unsigned)BD, 256, 0, stream>>>(cl, mcd0, nullptr, total_h, M_BR, 0);
  mcd_kernel<<<(unsigned)BD, 256, 0, stream>>>(cr, mcd0, nullptr, total_h, M_BR, 4);

  // ---- Level 1, right subtree -> branches 8..15 ----
  process_internal(nr, mcd1, P1, Ph1, ximf, ximf_h, hbs, sel, xsum0, xsum1,
                   t1, t1_h, cl, cr, stats, 0xC913u, stream);
  mcd_kernel<<<(unsigned)BD, 256, 0, stream>>>(cl, mcd0, nullptr, total_h, M_BR, 8);
  mcd_kernel<<<(unsigned)BD, 256, 0, stream>>>(cr, mcd0, nullptr, total_h, M_BR, 12);

  // ---- Per-branch forecast heads: Linear -> LeakyReLU -> Linear ----
  float* out = (float*)d_out;                           // [B,D,16,L] f32
  for (int m = 0; m < M_BR; ++m) {
    // hh = leaky(total[:,:,m,:] @ fc_w1[m]^T + fc_b1[m])  (f16 out, feeds GEMM2)
    launch_gemm(total_h + (long)m * L_DIM, (long)M_BR * L_DIM,
                fcw1_h + (long)m * WREC, fc_b1 + (long)m * H_DIM,
                nullptr, hh_h, H_DIM, (int)BD, H_DIM, L_DIM, /*leaky|f16*/ 3, stream);
    // out[:,:,m,:] = hh @ fc_w2[m]^T + fc_b2[m]
    launch_gemm(hh_h, H_DIM,
                fcw2_h + (long)m * WREC, fc_b2 + (long)m * L_DIM,
                out + (long)m * L_DIM, nullptr, (long)M_BR * L_DIM,
                (int)BD, L_DIM, H_DIM, 0, stream);
  }
}
